// SuperpositionAwarePrototypeMemory_69329362092330
// MI455X (gfx1250) — 8-bit matrix operand order repaired and checked by reading, compiled, not run on hardware
//
#include <hip/hip_runtime.h>
#include <hip/hip_bf16.h>

typedef int v8i __attribute__((ext_vector_type(8)));

#define NUM_CLASSES 1000
#define CPAD        1024
#define FEAT_DIM    256
#define NROWS       200000
#define BIMG        64
#define KLAB        100

#define WS_PT_BYTES   (CPAD * BIMG)
#define WS_CNT_OFF    WS_PT_BYTES
#define WS_CNT_BYTES  (CPAD * 4)
#define WS_SUM_OFF    (WS_CNT_OFF + WS_CNT_BYTES)
#define WS_SUM_BYTES  (NUM_CLASSES * FEAT_DIM * 4)
#define WS_TOTAL_WORDS ((WS_SUM_OFF + WS_SUM_BYTES) / 4)

__global__ void k_zero_ws(unsigned int* __restrict__ ws, int nwords) {
    int stride = gridDim.x * blockDim.x;
    for (int i = blockIdx.x * blockDim.x + threadIdx.x; i < nwords; i += stride)
        ws[i] = 0u;
}

__global__ void k_scatter(const int* __restrict__ lpi, unsigned char* __restrict__ Pt) {
    int idx = blockIdx.x * blockDim.x + threadIdx.x;
    if (idx >= BIMG * KLAB) return;
    int b   = idx / KLAB;
    int lab = lpi[idx];
    Pt[(size_t)lab * BIMG + b] = 1;
}

__global__ void k_feat_accum(const float* __restrict__ feats,
                             const int*   __restrict__ labels,
                             float* __restrict__ sums,
                             float* __restrict__ counts) {
    int wave = (blockIdx.x * blockDim.x + threadIdx.x) >> 5;
    int lane = threadIdx.x & 31;
    if (wave >= NROWS) return;

    const float4* row = (const float4*)(feats + (size_t)wave * FEAT_DIM);
    float4 a = row[lane];
    float4 b = row[lane + 32];

    float ss = a.x*a.x + a.y*a.y + a.z*a.z + a.w*a.w
             + b.x*b.x + b.y*b.y + b.z*b.z + b.w*b.w;
    #pragma unroll
    for (int off = 16; off > 0; off >>= 1)
        ss += __shfl_xor(ss, off, 32);

    float inv = 1.0f / fmaxf(sqrtf(ss), 1e-12f);

    int lab = labels[wave];
    float* s = sums + (size_t)lab * FEAT_DIM;
    atomicAdd(&s[4*lane + 0],       a.x * inv);
    atomicAdd(&s[4*lane + 1],       a.y * inv);
    atomicAdd(&s[4*lane + 2],       a.z * inv);
    atomicAdd(&s[4*lane + 3],       a.w * inv);
    atomicAdd(&s[128 + 4*lane + 0], b.x * inv);
    atomicAdd(&s[128 + 4*lane + 1], b.y * inv);
    atomicAdd(&s[128 + 4*lane + 2], b.z * inv);
    atomicAdd(&s[128 + 4*lane + 3], b.w * inv);
    if (lane == 0) atomicAdd(&counts[lab], 1.0f);
}

__global__ void k_finalize(const float* __restrict__ protos,
                           const int*   __restrict__ init_mask,
                           const int*   __restrict__ step_p,
                           const float* __restrict__ sums,
                           const float* __restrict__ counts,
                           float* __restrict__ out_protos,
                           int*   __restrict__ out_init) {
    int idx = blockIdx.x * blockDim.x + threadIdx.x;
    if (idx >= NUM_CLASSES * FEAT_DIM) return;
    int c = idx >> 8;

    float cnt  = counts[c];
    float mean = sums[idx] / fmaxf(cnt, 1.0f);

    float progress = fminf(1.0f, (float)(*step_p) / 2000.0f);
    float m = 0.99f + (0.999f - 0.99f) * progress;

    float p   = protos[idx];
    float ema = m * p + (1.0f - m) * mean;

    bool present = cnt > 0.0f;
    bool inited  = init_mask[c] > 0;
    out_protos[idx] = present ? (inited ? ema : mean) : p;

    if ((idx & 255) == 0)
        out_init[c] = (inited || present) ? 1 : 0;
}

__global__ void __launch_bounds__(128)
k_cooc_wmma(const unsigned char* __restrict__ Pt,
            const float* __restrict__ cooc_in,
            float* __restrict__ cooc_out) {
    int wave = (blockIdx.x * blockDim.x + threadIdx.x) >> 5;
    int lane = threadIdx.x & 31;
    int ti = wave >> 6;
    int tj = wave & 63;
    int ci0 = ti * 16, cj0 = tj * 16;

    int am     = lane & 15;
    int kbase  = (lane >> 4) * 32;
    const unsigned char* arow = Pt + (size_t)(ci0 + am) * 64 + kbase;
    uint2 c0 = *(const uint2*)(arow +  0);
    uint2 c1 = *(const uint2*)(arow +  8);
    uint2 c2 = *(const uint2*)(arow + 16);
    uint2 c3 = *(const uint2*)(arow + 24);
    v8i A;
    A[0] = (int)c0.x; A[1] = (int)c0.y;
    A[2] = (int)c1.x; A[3] = (int)c1.y;
    A[4] = (int)c2.x; A[5] = (int)c2.y;
    A[6] = (int)c3.x; A[7] = (int)c3.y;

    const unsigned char* brow = Pt + (size_t)(cj0 + (lane & 15)) * 64 + (lane >> 4) * 32;
    v8i B = *(const v8i*)brow;

    v8i C = {};
    v8i D = __builtin_amdgcn_wmma_i32_16x16x64_iu8(
         false, A,  false, B, C,
         false,  false);

    int col = cj0 + (lane & 15);
    int rbase = ci0 + ((lane >> 4) << 3);
    #pragma unroll
    for (int r = 0; r < 8; ++r) {
        int row = rbase + r;
        if (row < NUM_CLASSES && col < NUM_CLASSES) {
            size_t o = (size_t)row * NUM_CLASSES + col;
            float add = (row == col) ? 0.0f : (float)D[r];
            cooc_out[o] = cooc_in[o] + add;
        }
    }
}

extern "C" void kernel_launch(void* const* d_in, const int* in_sizes, int n_in,
                              void* d_out, int out_size, void* d_ws, size_t ws_size,
                              hipStream_t stream) {
    const float* features   = (const float*)d_in[0];
    const int*   labels     = (const int*)  d_in[1];
    const int*   lpi        = (const int*)  d_in[2];
    const float* prototypes = (const float*)d_in[3];
    const int*   init_mask  = (const int*)  d_in[4];
    const float* cooc_in    = (const float*)d_in[5];
    const int*   step_p     = (const int*)  d_in[6];

    unsigned char* ws_b   = (unsigned char*)d_ws;
    unsigned char* Pt     = ws_b;
    float*         counts = (float*)(ws_b + WS_CNT_OFF);
    float*         sums   = (float*)(ws_b + WS_SUM_OFF);

    float* out_protos = (float*)d_out;
    int*   out_init   = (int*)((float*)d_out + 256000);
    float* out_cooc   = (float*)d_out + 257000;

    k_zero_ws<<<512, 256, 0, stream>>>((unsigned int*)d_ws, WS_TOTAL_WORDS);

    k_scatter<<<(BIMG * KLAB + 255) / 256, 256, 0, stream>>>(lpi, Pt);

    k_feat_accum<<<(NROWS * 32) / 256, 256, 0, stream>>>(features, labels, sums, counts);

    k_finalize<<<(NUM_CLASSES * FEAT_DIM) / 256, 256, 0, stream>>>(
        prototypes, init_mask, step_p, sums, counts, out_protos, out_init);

    k_cooc_wmma<<<4096 / 4, 128, 0, stream>>>(Pt, cooc_in, out_cooc);
}
